// MultiHeadSelfAttention_45629732552827
// MI455X (gfx1250) — matrix operand order repaired and checked by reading, not compiled, not run on hardware
//
#include <hip/hip_runtime.h>
#include <hip/hip_bf16.h>

// MI455X / gfx1250: wave32, WMMA 16x16x32 bf16 -> f32 accumulate.
typedef __attribute__((ext_vector_type(16))) __bf16 v16bf;
typedef __attribute__((ext_vector_type(8)))  float  v8f;

#define NUMH   16
#define HDIM   64
#define DM     1024
#define BATCH  2
#define SEQ    2048
#define NTOK   (BATCH * SEQ)     // 4096

// softmax done in base-2: fold (1/sqrt(D)) * log2(e) into Q once at RoPE time
#define QSCALE (0.125f * 1.4426950408889634f)

static __device__ __forceinline__ __bf16 f2bf(float x) { return (__bf16)x; }

// pack two fp32 -> two bf16 in one dword
static __device__ __forceinline__ unsigned pack2bf(float x, float y) {
  union { __bf16 b[2]; unsigned u; } t;
  t.b[0] = (__bf16)x; t.b[1] = (__bf16)y;
  return t.u;
}

// 32-byte WMMA operand fragment viewed as two b128 chunks
union Frag16 {
  uint4 q[2];
  v16bf v;
};

// async global->LDS copy of 16 bytes per lane (gfx1250 ASYNCcnt path)
static __device__ __forceinline__ void async_ld_b128(const void* gptr, void* lptr) {
  unsigned lds_off = (unsigned)(size_t)lptr;        // low 32 bits = LDS offset
  unsigned long long ga = (unsigned long long)(size_t)gptr;
  asm volatile("global_load_async_to_lds_b128 %0, %1, off"
               :: "v"(lds_off), "v"(ga) : "memory");
}
static __device__ __forceinline__ void wait_async0() {
  asm volatile("s_wait_asynccnt 0" ::: "memory");
}

// ---------------------------------------------------------------------------
// Generic GEMM:  Out[m, n] = sum_k A[m, k] * W[n, k]     (A @ W^T)
// M = 4096, N = 1024, K = 1024.  Block tile 128(M) x 64(N) x 32(K),
// 256 threads = 8 waves arranged 4(M) x 2(N); each wave owns 32x32 = 2x2 WMMA tiles.
// ---------------------------------------------------------------------------
template <bool A_BF16, bool OUT_HEADMAJOR>
__global__ __launch_bounds__(256)
void gemm_wt(const void* __restrict__ Ain, const float* __restrict__ W,
             void* __restrict__ Out) {
  constexpr int TM = 128, TN = 64, TK = 32;
  constexpr int LDT = TK + 8;                 // 40 elems = 80 B rows (16B multiple)
  __shared__ __bf16 As[TM][LDT];
  __shared__ __bf16 Bs[TN][LDT];

  const int tid  = threadIdx.x;
  const int lane = tid & 31;
  const int wave = tid >> 5;                  // 0..7
  const int wm   = wave & 3;                  // 4 waves along M
  const int wn   = wave >> 2;                 // 2 waves along N
  const int half = lane >> 4;
  const int l16  = lane & 15;

  const int m0 = blockIdx.y * TM;
  const int n0 = blockIdx.x * TN;

  v8f zero = {};
  v8f acc[2][2];
#pragma unroll
  for (int i = 0; i < 2; ++i)
#pragma unroll
    for (int j = 0; j < 2; ++j) acc[i][j] = zero;

  for (int kt = 0; kt < DM / TK; ++kt) {
    const int k0 = kt * TK;

    // ---- stage A tile (128 x 32) into LDS as bf16 (vector stores) ----
    if constexpr (A_BF16) {
      const __bf16* X = (const __bf16*)Ain;
#pragma unroll
      for (int it = 0; it < 2; ++it) {
        int q = tid + it * 256;               // 0..511, 8 bf16 each
        int row = q >> 2;
        int c8  = (q & 3) << 3;
        uint4 raw = *(const uint4*)(X + (size_t)(m0 + row) * DM + k0 + c8);
        *(uint4*)&As[row][c8] = raw;          // ds_store_b128
      }
    } else {
      const float* X = (const float*)Ain;
#pragma unroll
      for (int it = 0; it < 4; ++it) {
        int q = tid + it * 256;               // 0..1023, float4 each
        int row = q >> 3;
        int c4  = (q & 7) << 2;
        float4 f = *(const float4*)(X + (size_t)(m0 + row) * DM + k0 + c4);
        uint2 pk;
        pk.x = pack2bf(f.x, f.y);
        pk.y = pack2bf(f.z, f.w);
        *(uint2*)&As[row][c4] = pk;           // ds_store_b64
      }
    }
    // ---- stage B tile: rows n0..n0+63 of W (fp32 -> bf16) ----
#pragma unroll
    for (int it = 0; it < 2; ++it) {
      int q = tid + it * 256;                 // 0..511
      int row = q >> 3;
      int c4  = (q & 7) << 2;
      float4 f = *(const float4*)(W + (size_t)(n0 + row) * DM + k0 + c4);
      uint2 pk;
      pk.x = pack2bf(f.x, f.y);
      pk.y = pack2bf(f.z, f.w);
      *(uint2*)&Bs[row][c4] = pk;
    }
    if (kt + 1 < DM / TK)
      __builtin_prefetch(W + (size_t)(n0 + (tid & 63)) * DM + k0 + TK, 0, 1);
    __syncthreads();

    // ---- fragments (2x ds_load_b128 each) ----
    v16bf a[2], b[2];
#pragma unroll
    for (int s = 0; s < 2; ++s) {
      const uint4* rp = (const uint4*)&As[wm * 32 + s * 16 + l16][0];
      Frag16 fa;
      fa.q[0] = rp[half];
      fa.q[1] = rp[2 + half];
      a[s] = fa.v;
    }
#pragma unroll
    for (int s = 0; s < 2; ++s) {
      const uint4* cp = (const uint4*)&Bs[wn * 32 + s * 16 + l16][0];
      Frag16 fb;
      fb.q[0] = cp[half];
      fb.q[1] = cp[2 + half];
      b[s] = fb.v;
    }
#pragma unroll
    for (int i = 0; i < 2; ++i)
#pragma unroll
      for (int j = 0; j < 2; ++j)
        acc[i][j] = __builtin_amdgcn_wmma_f32_16x16x32_bf16(
            false, a[i], false, b[j], (short)0, acc[i][j], false, false);
    __syncthreads();
  }

  // ---- epilogue: C layout lane = col, reg r = row (r + 8*half) ----
#pragma unroll
  for (int i = 0; i < 2; ++i) {
#pragma unroll
    for (int j = 0; j < 2; ++j) {
#pragma unroll
      for (int r = 0; r < 8; ++r) {
        int m = m0 + wm * 32 + i * 16 + r + half * 8;
        int o = n0 + wn * 32 + j * 16 + l16;
        float val = acc[i][j][r];
        if constexpr (OUT_HEADMAJOR) {
          __bf16* Ob = (__bf16*)Out;
          int bb = m / SEQ, srow = m & (SEQ - 1);
          int h = o >> 6, d = o & 63;
          Ob[(((size_t)bb * NUMH + h) * SEQ + srow) * HDIM + d] = f2bf(val);
        } else {
          ((float*)Out)[(size_t)m * DM + o] = val;
        }
      }
    }
  }
}

// ---------------------------------------------------------------------------
// RoPE, in-place on bf16 Q and K in [B, H, S, D] layout.
// Q additionally absorbs (1/sqrt(D)) * log2(e) so attention scores are
// directly in base-2 softmax units.
// ---------------------------------------------------------------------------
__global__ __launch_bounds__(256)
void rope_kernel(__bf16* __restrict__ Q, __bf16* __restrict__ K) {
  int idx = blockIdx.x * blockDim.x + threadIdx.x;     // < B*H*S*32 = 2^21
  int i  = idx & 31;
  int s  = (idx >> 5) & (SEQ - 1);
  int bh = idx >> 16;
  float invf = __expf(-(float)(2 * i) * (1.0f / 64.0f) * __logf(10000.0f));
  float ang  = (float)s * invf;
  float sn, cs;
  __sincosf(ang, &sn, &cs);
  size_t base = ((size_t)bh * SEQ + s) * HDIM;
  float q0 = (float)Q[base + i], q1 = (float)Q[base + i + 32];
  Q[base + i]      = f2bf((q0 * cs - q1 * sn) * QSCALE);
  Q[base + i + 32] = f2bf((q1 * cs + q0 * sn) * QSCALE);
  float k0 = (float)K[base + i], k1 = (float)K[base + i + 32];
  K[base + i]      = f2bf(k0 * cs - k1 * sn);
  K[base + i + 32] = f2bf(k1 * cs + k0 * sn);
}

// ---------------------------------------------------------------------------
// Flash-style attention per (b,h). Grid (S/64, B*H), 128 threads = 4 waves,
// each wave owns 16 query rows. K chunk copied via global_load_async_to_lds,
// V chunk transposed into Vt[d][key]. Softmax denominator computed on the
// matrix pipe via an all-ones B fragment (row_sum = P @ 1).
// ---------------------------------------------------------------------------
__global__ __launch_bounds__(128)
void attn_kernel(const __bf16* __restrict__ Q, const __bf16* __restrict__ K,
                 const __bf16* __restrict__ V, __bf16* __restrict__ Ctx) {
  constexpr int BN  = 64;
  constexpr int LKV = HDIM + 8;               // 72 elems = 144 B (16B multiple)
  __shared__ __bf16 Ks[BN][LKV];              // Ks[key][d]
  __shared__ __bf16 Vt[HDIM][BN + 8];         // Vt[d][key], stride 144 B
  __shared__ __bf16 Ps[4][16][BN];            // per-wave P, stride 128 B

  const int tid  = threadIdx.x;
  const int lane = tid & 31;
  const int wave = tid >> 5;                  // 0..3
  const int half = lane >> 4;
  const int l16  = lane & 15;

  const int bh    = blockIdx.y;               // b*H + h
  const int q0row = blockIdx.x * 64 + wave * 16;

  // Q fragments: two 16x32 A-frags covering D=64, loaded as global b128 pairs.
  v16bf aq[2];
  const uint4* qp = (const uint4*)(Q + ((size_t)bh * SEQ + q0row + l16) * HDIM);
#pragma unroll
  for (int ks = 0; ks < 2; ++ks) {
    Frag16 fa;
    fa.q[0] = qp[ks * 4 + half];
    fa.q[1] = qp[ks * 4 + 2 + half];
    aq[ks] = fa.v;
  }

  // all-ones B fragment: every output column of (P @ ones) = row sum of P
  v16bf ones;
#pragma unroll
  for (int e = 0; e < 16; ++e) ones[e] = f2bf(1.0f);

  v8f zero = {};
  v8f acco[4];                                 // context accumulators
  v8f accd = zero;                             // softmax denominator accumulator
#pragma unroll
  for (int dt = 0; dt < 4; ++dt) acco[dt] = zero;
  float mrow[8];
#pragma unroll
  for (int r = 0; r < 8; ++r) mrow[r] = -3.0e38f;

  for (int j = 0; j < SEQ / BN; ++j) {
    // ---- K chunk: async global->LDS copy (ASYNCcnt) ----
#pragma unroll
    for (int it = 0; it < 4; ++it) {
      int qq  = tid + it * 128;               // 0..511, 8 bf16 each
      int row = qq >> 3;
      int c8  = (qq & 7) << 3;
      async_ld_b128(K + ((size_t)bh * SEQ + j * BN + row) * HDIM + c8,
                    &Ks[row][c8]);
    }
    // ---- V chunk: transpose into Vt[d][key] with packed b32 stores ----
#pragma unroll
    for (int it = 0; it < 2; ++it) {
      int idx = tid + it * 128;               // 0..255
      int kp  = idx >> 3;                     // key pair 0..31
      int c8  = (idx & 7) << 3;               // d group
      size_t g0 = ((size_t)bh * SEQ + j * BN + 2 * kp) * HDIM + c8;
      union { uint4 q; unsigned short s[8]; } ua, ub;
      ua.q = *(const uint4*)(V + g0);
      ub.q = *(const uint4*)(V + g0 + HDIM);
#pragma unroll
      for (int e = 0; e < 8; ++e) {
        unsigned valp = (unsigned)ua.s[e] | ((unsigned)ub.s[e] << 16);
        *(unsigned*)&Vt[c8 + e][2 * kp] = valp;  // ds_store_b32
      }
    }
    wait_async0();
    __syncthreads();

    // ---- S = Q K^T : 16 rows x 64 keys (scores already in log2 units) ----
    v8f accs[4];
#pragma unroll
    for (int nt = 0; nt < 4; ++nt) accs[nt] = zero;
#pragma unroll
    for (int nt = 0; nt < 4; ++nt) {
      const uint4* cp = (const uint4*)&Ks[nt * 16 + l16][0];   // B col = key
#pragma unroll
      for (int ks = 0; ks < 2; ++ks) {
        Frag16 fb;
        fb.q[0] = cp[ks * 4 + half];
        fb.q[1] = cp[ks * 4 + 2 + half];
        accs[nt] = __builtin_amdgcn_wmma_f32_16x16x32_bf16(
            false, aq[ks], false, fb.v, (short)0, accs[nt], false, false);
      }
    }

    // ---- online softmax: max via shfl, rescale acc by alpha = 2^(m_old-m_new)
#pragma unroll
    for (int r = 0; r < 8; ++r) {
      float mx = accs[0][r];
#pragma unroll
      for (int nt = 1; nt < 4; ++nt) mx = fmaxf(mx, accs[nt][r]);
#pragma unroll
      for (int off = 8; off >= 1; off >>= 1) mx = fmaxf(mx, __shfl_xor(mx, off, 32));
      float nm    = fmaxf(mrow[r], mx);
      float alpha = exp2f(mrow[r] - nm);
      mrow[r] = nm;
      accd[r] *= alpha;
#pragma unroll
      for (int dt = 0; dt < 4; ++dt) acco[dt][r] *= alpha;
    }
    // ---- P = 2^(S - m), stash in per-wave LDS for the A-layout transpose ----
#pragma unroll
    for (int nt = 0; nt < 4; ++nt) {
#pragma unroll
      for (int r = 0; r < 8; ++r) {
        float p = exp2f(accs[nt][r] - mrow[r]);
        Ps[wave][r + half * 8][nt * 16 + l16] = f2bf(p);
      }
    }

    // ---- context += P @ V ; denominator += P @ ones (matrix pipe) ----
#pragma unroll
    for (int ks2 = 0; ks2 < 2; ++ks2) {
      const uint4* pp = (const uint4*)&Ps[wave][l16][0];
      Frag16 fa;
      fa.q[0] = pp[ks2 * 4 + half];
      fa.q[1] = pp[ks2 * 4 + 2 + half];
      accd = __builtin_amdgcn_wmma_f32_16x16x32_bf16(
          false, fa.v, false, ones, (short)0, accd, false, false);
#pragma unroll
      for (int dt = 0; dt < 4; ++dt) {
        const uint4* vp = (const uint4*)&Vt[dt * 16 + l16][0]; // B col = d
        Frag16 fb;
        fb.q[0] = vp[ks2 * 4 + half];
        fb.q[1] = vp[ks2 * 4 + 2 + half];
        acco[dt] = __builtin_amdgcn_wmma_f32_16x16x32_bf16(
            false, fa.v, false, fb.v, (short)0, acco[dt], false, false);
      }
    }
    __syncthreads();
  }

  // ---- normalize and scatter context to [B, S, H*D] bf16 ----
  const int bb = bh >> 4;
  const int h  = bh & 15;
#pragma unroll
  for (int r = 0; r < 8; ++r) {
    int   srow = q0row + r + half * 8;
    float invd = 1.0f / accd[r];
#pragma unroll
    for (int dt = 0; dt < 4; ++dt) {
      int d = dt * 16 + l16;
      Ctx[((size_t)bb * SEQ + srow) * DM + h * HDIM + d] = f2bf(acco[dt][r] * invd);
    }
  }
}

// ---------------------------------------------------------------------------
extern "C" void kernel_launch(void* const* d_in, const int* in_sizes, int n_in,
                              void* d_out, int out_size, void* d_ws, size_t ws_size,
                              hipStream_t stream) {
  (void)in_sizes; (void)n_in; (void)out_size; (void)ws_size;
  const float* hs = (const float*)d_in[0];
  // d_in[1] = attention_mask (all true) -> no-op for the math
  const float* Wq = (const float*)d_in[2];
  const float* Wk = (const float*)d_in[3];
  const float* Wv = (const float*)d_in[4];
  const float* Wo = (const float*)d_in[5];

  const size_t perBuf = (size_t)BATCH * NUMH * SEQ * HDIM;   // 4.19M elements
  __bf16* Qb = (__bf16*)d_ws;
  __bf16* Kb = Qb + perBuf;
  __bf16* Vb = Kb + perBuf;
  __bf16* Cb = Vb + perBuf;        // context, [B, S, Dm] bf16

  dim3 gGemm(DM / 64, NTOK / 128);
  gemm_wt<false, true><<<gGemm, 256, 0, stream>>>(hs, Wq, Qb);
  gemm_wt<false, true><<<gGemm, 256, 0, stream>>>(hs, Wk, Kb);
  gemm_wt<false, true><<<gGemm, 256, 0, stream>>>(hs, Wv, Vb);

  int ropeN = BATCH * NUMH * SEQ * 32;
  rope_kernel<<<ropeN / 256, 256, 0, stream>>>(Qb, Kb);

  dim3 gAttn(SEQ / 64, BATCH * NUMH);
  attn_kernel<<<gAttn, 128, 0, stream>>>(Qb, Kb, Vb, Cb);

  gemm_wt<true, false><<<gGemm, 256, 0, stream>>>(Cb, Wo, (float*)d_out);
}
